// VNEncoder_47768626266590
// MI455X (gfx1250) — matrix operand order repaired and checked by reading, not compiled, not run on hardware
//
#include <hip/hip_runtime.h>
#include <hip/hip_bf16.h>
#include <stdint.h>

// ---------------------------------------------------------------------------
// VN encoder for MI455X (gfx1250, wave32).  B=4, N=1024, K=20.
// theta_net GEMMs (t1/t2/t3) run on V_WMMA_F32_16X16X32_BF16 (fp32 accum)
// with LDS double-buffering filled via GLOBAL_LOAD_ASYNC_TO_LDS_B128.
// ---------------------------------------------------------------------------

#define NBATCH 4
#define NPTS   1024
#define KNBR   20
#define TOTF   169      // concatenated FPN channels

typedef __bf16 BF16;
typedef __attribute__((ext_vector_type(16))) __bf16 v16bf;
typedef __attribute__((ext_vector_type(8)))  __bf16 v8bf;
typedef __attribute__((ext_vector_type(8)))  float  v8f;

__device__ inline BF16 f2bf(float x) {
    union { float f; unsigned u; } a; a.f = x;
    unsigned r = a.u + 0x7FFFu + ((a.u >> 16) & 1u);   // round-to-nearest-even
    unsigned short s = (unsigned short)(r >> 16);
    union { unsigned short s; BF16 b; } o; o.s = s; return o.b;
}

__device__ inline void wait_async0() {
#if __has_builtin(__builtin_amdgcn_s_wait_asynccnt)
    __builtin_amdgcn_s_wait_asynccnt(0);
#else
    asm volatile("s_wait_asynccnt 0x0" ::: "memory");
#endif
}

// ------------------------------ utility ------------------------------------
__global__ void zero_u32(unsigned* __restrict__ p, size_t n) {
    size_t i = (size_t)blockIdx.x * blockDim.x + threadIdx.x;
    size_t s = (size_t)gridDim.x * blockDim.x;
    for (; i < n; i += s) p[i] = 0u;
}

// x [B,N,3] -> xt [B*N, 32] (point-major, zero padded) for KNN stage 1
__global__ void pack_x(const float* __restrict__ x, float* __restrict__ xt) {
    int t = blockIdx.x * 256 + threadIdx.x;
    if (t >= NBATCH * NPTS) return;
    const float* xp = x + (size_t)t * 3;
    float* o = xt + (size_t)t * 32;
    o[0] = xp[0]; o[1] = xp[1]; o[2] = xp[2];
    for (int d = 3; d < 32; ++d) o[d] = 0.f;
}

// x [B,N,3] -> x0 [B,1,3,N] channel-major
__global__ void x0fill(const float* __restrict__ x, float* __restrict__ x0) {
    int t = blockIdx.x * 256 + threadIdx.x;
    if (t >= NBATCH * NPTS) return;
    int b = t / NPTS, n = t % NPTS;
    for (int c = 0; c < 3; ++c)
        x0[((size_t)b * 3 + c) * NPTS + n] = x[(size_t)t * 3 + c];
}

// fcat sub-range [B][Cch][3][N] -> xt [B*N][Dpad] (D = Cch*3, zero padded)
__global__ void feat_to_xt(const float* __restrict__ f, int srcOff, int srcTot,
                           int Cch, float* __restrict__ xt, int Dpad) {
    int t = blockIdx.x * 256 + threadIdx.x;
    if (t >= NBATCH * NPTS) return;
    int b = t / NPTS, n = t % NPTS;
    float* o = xt + (size_t)t * Dpad;
    int D = Cch * 3;
    for (int d = 0; d < Dpad; ++d)
        o[d] = (d < D)
             ? f[(((size_t)b * srcTot + srcOff + d / 3) * 3 + (d % 3)) * NPTS + n]
             : 0.f;
}

// ------------------------------ KNN ----------------------------------------
// brute-force top-20 of -||xi-xj||^2 (self included, like jax top_k).
__global__ void __launch_bounds__(128)
knn_topk(const float* __restrict__ xt, int Dpad, int* __restrict__ idxo) {
    __shared__ float cs[32][33];
    const int tid = threadIdx.x;
    const int bpb = NPTS / 128;
    const int b  = blockIdx.x / bpb;
    const int qi = (blockIdx.x % bpb) * 128 + tid;
    const float* base = xt + (size_t)b * NPTS * Dpad;
    float bestd[KNBR]; int besti[KNBR];
#pragma unroll
    for (int s = 0; s < KNBR; ++s) { bestd[s] = -3.0e38f; besti[s] = 0; }
    const int nchunk = Dpad >> 5;
    for (int ct = 0; ct < NPTS / 32; ++ct) {
        float acc[32];
#pragma unroll
        for (int c = 0; c < 32; ++c) acc[c] = 0.f;
        for (int ch = 0; ch < nchunk; ++ch) {
            __syncthreads();
            for (int t = tid; t < 1024; t += 128) {
                int cc = t >> 5, dd = t & 31;
                cs[cc][dd] = base[(size_t)(ct * 32 + cc) * Dpad + ch * 32 + dd];
            }
            __syncthreads();
            float q[32];
#pragma unroll
            for (int j = 0; j < 32; ++j) q[j] = base[(size_t)qi * Dpad + ch * 32 + j];
            for (int c = 0; c < 32; ++c) {
                float a = acc[c];
#pragma unroll
                for (int j = 0; j < 32; ++j) { float d = q[j] - cs[c][j]; a += d * d; }
                acc[c] = a;
            }
        }
        for (int c = 0; c < 32; ++c) {
            float vd = -acc[c]; int vi = ct * 32 + c;
#pragma unroll
            for (int s = 0; s < KNBR; ++s) {
                if (vd > bestd[s]) {
                    float td = bestd[s]; int ti = besti[s];
                    bestd[s] = vd; besti[s] = vi; vd = td; vi = ti;
                }
            }
        }
    }
    int* orow = idxo + ((size_t)b * NPTS + qi) * KNBR;
#pragma unroll
    for (int s = 0; s < KNBR; ++s) orow[s] = besti[s];
}

// -------------------- edge-feature VN blocks (c1..c4) ----------------------
#define FEAT(b,i,c,j) feat[(((size_t)(b)*srcTot + srcOff + (i))*3 + (c))*NPTS + (j)]

template<int C, int O, bool CROSS>
__global__ void __launch_bounds__(128)
vn_edge_passA(const float* __restrict__ feat, int srcOff, int srcTot,
              const int* __restrict__ idxp, const float* __restrict__ Wf,
              float* __restrict__ stats) {
    constexpr int CP = CROSS ? 3 * C : 2 * C;
    __shared__ float e[KNBR][CP][3];
    __shared__ float ls1[O], ls2[O];
    const int b = blockIdx.x / NPTS, n = blockIdx.x % NPTS;
    const int tid = threadIdx.x;
    for (int t = tid; t < O; t += 128) { ls1[t] = 0.f; ls2[t] = 0.f; }
    const int* row = idxp + ((size_t)b * NPTS + n) * KNBR;
    for (int t = tid; t < KNBR * CP * 3; t += 128) {
        int kk = t / (CP * 3); int r = t % (CP * 3); int i = r / 3; int c = r % 3;
        int j = row[kk];
        float v;
        if (i < C) v = FEAT(b, i, c, j) - FEAT(b, i, c, n);
        else if (i < 2 * C) v = FEAT(b, i - C, c, n);
        else {
            int ii = i - 2 * C;
            int c1 = (c + 1) % 3, c2 = (c + 2) % 3;
            v = FEAT(b, ii, c1, j) * FEAT(b, ii, c2, n)
              - FEAT(b, ii, c2, j) * FEAT(b, ii, c1, n);
        }
        e[kk][i][c] = v;
    }
    __syncthreads();
    for (int t = tid; t < O * KNBR; t += 128) {
        int o = t / KNBR, kk = t % KNBR;
        const float* wf = Wf + o * CP;
        float q0 = 0, q1 = 0, q2 = 0;
        for (int i = 0; i < CP; ++i) {
            float w = wf[i];
            q0 += w * e[kk][i][0]; q1 += w * e[kk][i][1]; q2 += w * e[kk][i][2];
        }
        float nrm = sqrtf(q0 * q0 + q1 * q1 + q2 * q2) + 1e-6f;
        atomicAdd(&ls1[o], nrm); atomicAdd(&ls2[o], nrm * nrm);
    }
    __syncthreads();
    for (int t = tid; t < O; t += 128) {
        atomicAdd(&stats[t], ls1[t]); atomicAdd(&stats[O + t], ls2[t]);
    }
}

template<int C, int O, bool CROSS>
__global__ void __launch_bounds__(128)
vn_edge_passB(const float* __restrict__ feat, int srcOff, int srcTot,
              const int* __restrict__ idxp,
              const float* __restrict__ Wf, const float* __restrict__ Wd,
              const float* __restrict__ gam, const float* __restrict__ bet,
              const float* __restrict__ stats,
              float* __restrict__ out, int chOff, float ns) {
    constexpr int CP = CROSS ? 3 * C : 2 * C;
    __shared__ float e[KNBR][CP][3];
    const int b = blockIdx.x / NPTS, n = blockIdx.x % NPTS;
    const int tid = threadIdx.x;
    const int* row = idxp + ((size_t)b * NPTS + n) * KNBR;
    for (int t = tid; t < KNBR * CP * 3; t += 128) {
        int kk = t / (CP * 3); int r = t % (CP * 3); int i = r / 3; int c = r % 3;
        int j = row[kk];
        float v;
        if (i < C) v = FEAT(b, i, c, j) - FEAT(b, i, c, n);
        else if (i < 2 * C) v = FEAT(b, i - C, c, n);
        else {
            int ii = i - 2 * C;
            int c1 = (c + 1) % 3, c2 = (c + 2) % 3;
            v = FEAT(b, ii, c1, j) * FEAT(b, ii, c2, n)
              - FEAT(b, ii, c2, j) * FEAT(b, ii, c1, n);
        }
        e[kk][i][c] = v;
    }
    __syncthreads();
    const float cntInv = 1.0f / (float)(NBATCH * NPTS * KNBR);
    for (int o = tid; o < O; o += 128) {
        float mu  = stats[o] * cntInv;
        float var = stats[O + o] * cntInv - mu * mu;
        float inv = rsqrtf(var + 1e-5f);
        float ga = gam[o], be = bet[o];
        const float* wf = Wf + o * CP;
        const float* wd = Wd + o * CP;
        float a0 = 0, a1 = 0, a2 = 0;
        for (int kk = 0; kk < KNBR; ++kk) {
            float q0 = 0, q1 = 0, q2 = 0, d0 = 0, d1 = 0, d2 = 0;
            for (int i = 0; i < CP; ++i) {
                float w = wf[i], wdd = wd[i];
                float e0 = e[kk][i][0], e1 = e[kk][i][1], e2 = e[kk][i][2];
                q0 += w * e0; q1 += w * e1; q2 += w * e2;
                d0 += wdd * e0; d1 += wdd * e1; d2 += wdd * e2;
            }
            float nrm = sqrtf(q0 * q0 + q1 * q1 + q2 * q2) + 1e-6f;
            float nbn = (nrm - mu) * inv * ga + be;
            float sc = nbn / nrm;
            q0 *= sc; q1 *= sc; q2 *= sc;
            float dot = q0 * d0 + q1 * d1 + q2 * d2;
            float dd  = d0 * d0 + d1 * d1 + d2 * d2 + 1e-6f;
            float f = (dot >= 0.f) ? 0.f : dot / dd;
            a0 += ns * q0 + (1.f - ns) * (q0 - f * d0);
            a1 += ns * q1 + (1.f - ns) * (q1 - f * d1);
            a2 += ns * q2 + (1.f - ns) * (q2 - f * d2);
        }
        const float ki = 1.0f / (float)KNBR;
        out[(((size_t)b * TOTF + chOff + o) * 3 + 0) * NPTS + n] = a0 * ki;
        out[(((size_t)b * TOTF + chOff + o) * 3 + 1) * NPTS + n] = a1 * ki;
        out[(((size_t)b * TOTF + chOff + o) * 3 + 2) * NPTS + n] = a2 * ki;
    }
}

// -------------------- dense VN blocks (c5 / s1 / s2) -----------------------
// hmode: input channel i<170 -> g[b,i,c,n], i>=170 -> z[b,i-170,c] broadcast.
__global__ void __launch_bounds__(256)
vn_dense_passA(const float* __restrict__ src, const float* __restrict__ zsrc,
               int hmode, int Cin, int srcC, int O, int sharedD,
               const float* __restrict__ Wf, const float* __restrict__ Wd,
               float* __restrict__ qb, float* __restrict__ db,
               float* __restrict__ stats) {
    __shared__ float wfr[340], wdr[340];
    __shared__ float r1[256], r2[256];
    const int b = blockIdx.x / O, o = blockIdx.x % O;
    const int tid = threadIdx.x;
    for (int i = tid; i < Cin; i += 256) {
        wfr[i] = Wf[(size_t)o * Cin + i];
        wdr[i] = sharedD ? Wd[i] : Wd[(size_t)o * Cin + i];
    }
    __syncthreads();
    float s1 = 0, s2 = 0;
    for (int n = tid; n < NPTS; n += 256) {
        float q0 = 0, q1 = 0, q2 = 0, d0 = 0, d1 = 0, d2 = 0;
        for (int i = 0; i < Cin; ++i) {
            float x0, x1, x2;
            if (hmode && i >= 170) {
                const float* zp = zsrc + ((size_t)b * 170 + (i - 170)) * 3;
                x0 = zp[0]; x1 = zp[1]; x2 = zp[2];
            } else {
                const float* xp = src + (((size_t)b * srcC + i) * 3) * NPTS + n;
                x0 = xp[0]; x1 = xp[NPTS]; x2 = xp[2 * NPTS];
            }
            float w = wfr[i], wd = wdr[i];
            q0 += w * x0; q1 += w * x1; q2 += w * x2;
            d0 += wd * x0; d1 += wd * x1; d2 += wd * x2;
        }
        float* qp = qb + (((size_t)b * O + o) * 3) * NPTS + n;
        qp[0] = q0; qp[NPTS] = q1; qp[2 * NPTS] = q2;
        if (!sharedD) {
            float* dp = db + (((size_t)b * O + o) * 3) * NPTS + n;
            dp[0] = d0; dp[NPTS] = d1; dp[2 * NPTS] = d2;
        } else if (o == 0) {
            float* dp = db + ((size_t)b * 3) * NPTS + n;
            dp[0] = d0; dp[NPTS] = d1; dp[2 * NPTS] = d2;
        }
        float nrm = sqrtf(q0 * q0 + q1 * q1 + q2 * q2) + 1e-6f;
        s1 += nrm; s2 += nrm * nrm;
    }
    r1[tid] = s1; r2[tid] = s2; __syncthreads();
    for (int st = 128; st > 0; st >>= 1) {
        if (tid < st) { r1[tid] += r1[tid + st]; r2[tid] += r2[tid + st]; }
        __syncthreads();
    }
    if (tid == 0) { atomicAdd(&stats[o], r1[0]); atomicAdd(&stats[O + o], r2[0]); }
}

__global__ void __launch_bounds__(256)
vn_dense_passB(const float* __restrict__ qb, const float* __restrict__ db,
               const float* __restrict__ stats,
               const float* __restrict__ gam, const float* __restrict__ bet,
               float* __restrict__ dst, int O, int sharedD, float ns, int total) {
    int t = blockIdx.x * 256 + threadIdx.x;
    int stride = gridDim.x * 256;
    const float cntInv = 1.0f / (float)(NBATCH * NPTS);
    for (; t < total; t += stride) {
        int n = t % NPTS; int r = t / NPTS; int o = r % O; int b = r / O;
        float mu  = stats[o] * cntInv;
        float var = stats[O + o] * cntInv - mu * mu;
        float inv = rsqrtf(var + 1e-5f);
        const float* qp = qb + (((size_t)b * O + o) * 3) * NPTS + n;
        float q0 = qp[0], q1 = qp[NPTS], q2 = qp[2 * NPTS];
        const float* dp = sharedD ? db + ((size_t)b * 3) * NPTS + n
                                  : db + (((size_t)b * O + o) * 3) * NPTS + n;
        float d0 = dp[0], d1 = dp[NPTS], d2 = dp[2 * NPTS];
        float nrm = sqrtf(q0 * q0 + q1 * q1 + q2 * q2) + 1e-6f;
        float nbn = (nrm - mu) * inv * gam[o] + bet[o];
        float sc = nbn / nrm;
        q0 *= sc; q1 *= sc; q2 *= sc;
        float dot = q0 * d0 + q1 * d1 + q2 * d2;
        float dd  = d0 * d0 + d1 * d1 + d2 * d2 + 1e-6f;
        float f = (dot >= 0.f) ? 0.f : dot / dd;
        float* op = dst + (((size_t)b * O + o) * 3) * NPTS + n;
        op[0]        = ns * q0 + (1.f - ns) * (q0 - f * d0);
        op[NPTS]     = ns * q1 + (1.f - ns) * (q1 - f * d1);
        op[2 * NPTS] = ns * q2 + (1.f - ns) * (q2 - f * d2);
    }
}

// ------------------------ small einsums / pooling --------------------------
__global__ void zmean_kernel(const float* __restrict__ g, float* __restrict__ zb,
                             float* __restrict__ zout) {
    int t = blockIdx.x * 256 + threadIdx.x;
    if (t >= NBATCH * 170 * 3) return;
    const float* row = g + (size_t)t * NPTS;
    float s = 0;
    for (int n = 0; n < NPTS; ++n) s += row[n];
    float v = s * (1.0f / NPTS);
    zb[t] = v; zout[t] = v;
}

__global__ void z0_kernel(const float* __restrict__ s, const float* __restrict__ slin,
                          float* __restrict__ z0) {
    int t = blockIdx.x * 256 + threadIdx.x;
    if (t >= NBATCH * 9 * NPTS) return;
    int n = t % NPTS; int r = t / NPTS; int k = r % 3; r /= 3; int j = r % 3; int b = r / 3;
    float a = 0;
    for (int i = 0; i < 85; ++i)
        a += slin[j * 85 + i] * s[(((size_t)b * 85 + i) * 3 + k) * NPTS + n];
    z0[(((size_t)b * 3 + j) * 3 + k) * NPTS + n] = a;
}

// xs[b, i*3+k, n] = sum_j h[b,i,j,n]*z0[b,j,k,n]  -> bf16, transposed [m=b*N+n][1024]
__global__ void xs_kernel(const float* __restrict__ g, const float* __restrict__ z,
                          const float* __restrict__ z0, BF16* __restrict__ XT) {
    int t = blockIdx.x * 256 + threadIdx.x;
    if (t >= NBATCH * 340 * NPTS) return;
    int n = t % NPTS; int r = t / NPTS; int i = r % 340; int b = r / 340;
    float h0, h1, h2;
    if (i < 170) {
        const float* gp = g + (((size_t)b * 170 + i) * 3) * NPTS + n;
        h0 = gp[0]; h1 = gp[NPTS]; h2 = gp[2 * NPTS];
    } else {
        const float* zp = z + ((size_t)b * 170 + (i - 170)) * 3;
        h0 = zp[0]; h1 = zp[1]; h2 = zp[2];
    }
    BF16* orow = XT + ((size_t)(b * NPTS + n)) * 1024 + i * 3;
    for (int k = 0; k < 3; ++k) {
        float v = h0 * z0[(((size_t)b * 3 + 0) * 3 + k) * NPTS + n]
                + h1 * z0[(((size_t)b * 3 + 1) * 3 + k) * NPTS + n]
                + h2 * z0[(((size_t)b * 3 + 2) * 3 + k) * NPTS + n];
        orow[k] = f2bf(v);
    }
}

// fp32 weights -> bf16, zero-padded to [Opad][Ipad]
__global__ void cvt_w(const float* __restrict__ W, BF16* __restrict__ out,
                      int O, int I, int Opad, int Ipad) {
    size_t t = (size_t)blockIdx.x * 256 + threadIdx.x;
    size_t total = (size_t)Opad * Ipad;
    size_t stride = (size_t)gridDim.x * 256;
    for (; t < total; t += stride) {
        int o = (int)(t / Ipad), i = (int)(t % Ipad);
        float v = (o < O && i < I) ? W[(size_t)o * I + i] : 0.f;
        out[t] = f2bf(v);
    }
}

// ----------------------------- WMMA GEMM -----------------------------------
// D[o][m] = sum_k A[o][k] * X[m][k]   (A: bf16 row-major [Opad][Kpad],
// X: bf16 K-major activations [M][Kpad]).
// Block = 8 waves -> 128x128 tile; wave -> 32x64 (2x4 WMMA tiles).
// K-step 32, LDS double-buffered via async global->LDS b128 copies.
union FragAB { v16bf v; v8bf h[2]; };

__device__ inline void store_relu_bf16T(v8f a, BF16* outT, int Opad, int col, int ob) {
    v8bf o;
#pragma unroll
    for (int v = 0; v < 8; ++v) { float f = a[v]; f = f > 0.f ? f : 0.f; o[v] = f2bf(f); }
    *(v8bf*)(outT + (size_t)col * Opad + ob) = o;
}

__device__ inline void store_f32_bias(v8f a, float* outF, const float* bias,
                                      int Oreal, int col, int ob) {
    int b = col >> 10; int n = col & (NPTS - 1);
#pragma unroll
    for (int v = 0; v < 8; ++v) {
        int o = ob + v;
        if (o < Oreal) outF[((size_t)b * Oreal + o) * NPTS + n] = a[v] + bias[o];
    }
}

// async-fill one 128x32 bf16 tile (8KB) from row-major [rows][ldK] source.
// 256 threads x 2 chunks of 16B.  ldsOff = byte offset of tile base.
__device__ inline void async_fill_tile(const BF16* __restrict__ gbase, int ldK,
                                       int row0, int kk, uint32_t ldsOff, int tid) {
#pragma unroll
    for (int u = 0; u < 2; ++u) {
        int chunk = tid + u * 256;              // 0..511
        int r = chunk >> 2, c = chunk & 3;      // r: 0..127, c: 0..3 (8 bf16 each)
        const BF16* g = gbase + (size_t)(row0 + r) * ldK + kk + c * 8;
        uint32_t l = ldsOff + (uint32_t)(r * 32 + c * 8) * 2;
        asm volatile("global_load_async_to_lds_b128 %0, %1, off"
                     :: "v"(l), "v"(g) : "memory");
    }
}

__global__ void __launch_bounds__(256)
gemm_bf16(const BF16* __restrict__ A, const BF16* __restrict__ Bm,
          int Opad, int Kpad, int M, int mode,
          BF16* __restrict__ outT, float* __restrict__ outF,
          const float* __restrict__ bias, int Oreal) {
    // LDS: A buffers [2][128][32], B buffers [2][128][32] (bf16) = 32KB
    __shared__ __align__(16) BF16 lds[16384];
    const uint32_t lbase = (uint32_t)(uintptr_t)&lds[0];
    const int tid  = threadIdx.x;
    const int lane = tid & 31;
    const int wid  = tid >> 5;          // 0..7
    const int rg = wid >> 1;            // wave row group 0..3  (32 rows each)
    const int cg = wid & 1;             // wave col group 0..1  (64 cols each)
    const int mb = M >> 7;              // col blocks
    const int br = blockIdx.x / mb, bc = blockIdx.x % mb;
    const int row0 = br * 128, col0 = bc * 128;
    const int lr = lane & 15, hi = lane >> 4;
    const int NK = Kpad >> 5;

    v8f acc[2][4];
#pragma unroll
    for (int ti = 0; ti < 2; ++ti)
#pragma unroll
        for (int tj = 0; tj < 4; ++tj) acc[ti][tj] = (v8f){};

    // prologue: fill buffer 0
    async_fill_tile(A,  Kpad, row0, 0, lbase + 0,     tid);
    async_fill_tile(Bm, Kpad, col0, 0, lbase + 16384, tid);
    wait_async0();
    __syncthreads();

    for (int ki = 0; ki < NK; ++ki) {
        const int cur = ki & 1;
        if (ki + 1 < NK) {
            async_fill_tile(A,  Kpad, row0, (ki + 1) * 32,
                            lbase + (cur ^ 1) * 8192, tid);
            async_fill_tile(Bm, Kpad, col0, (ki + 1) * 32,
                            lbase + 16384 + (cur ^ 1) * 8192, tid);
        }
        const BF16* lA = lds + cur * 4096;
        const BF16* lB = lds + 8192 + cur * 4096;

        FragAB af[2], bfrag[4];
#pragma unroll
        for (int ti = 0; ti < 2; ++ti) {
            const BF16* p = lA + ((rg * 32 + ti * 16 + lr) * 32 + hi * 8);
            af[ti].h[0] = *(const v8bf*)p;
            af[ti].h[1] = *(const v8bf*)(p + 16);
        }
#pragma unroll
        for (int tj = 0; tj < 4; ++tj) {
            const BF16* p = lB + ((cg * 64 + tj * 16 + lr) * 32 + hi * 8);
            bfrag[tj].h[0] = *(const v8bf*)p;
            bfrag[tj].h[1] = *(const v8bf*)(p + 16);
        }
#pragma unroll
        for (int ti = 0; ti < 2; ++ti)
#pragma unroll
            for (int tj = 0; tj < 4; ++tj)
                acc[ti][tj] = __builtin_amdgcn_wmma_f32_16x16x32_bf16(
                    false, af[ti].v, false, bfrag[tj].v, (short)0,
                    acc[ti][tj], false, false);

        wait_async0();       // next buffer's fills (this wave) complete
        __syncthreads();     // all waves done reading cur + their fills
    }

#pragma unroll
    for (int ti = 0; ti < 2; ++ti)
#pragma unroll
        for (int tj = 0; tj < 4; ++tj) {
            const int col = col0 + cg * 64 + tj * 16 + lr;
            const int ob  = row0 + rg * 32 + ti * 16 + hi * 8;
            if (mode == 0) store_relu_bf16T(acc[ti][tj], outT, Opad, col, ob);
            else           store_f32_bias(acc[ti][tj], outF, bias, Oreal, col, ob);
        }
}

// ---------------------------------------------------------------------------
extern "C" void kernel_launch(void* const* d_in, const int* in_sizes, int n_in,
                              void* d_out, int out_size, void* d_ws, size_t ws_size,
                              hipStream_t stream) {
    (void)in_sizes; (void)n_in; (void)out_size; (void)ws_size;
    // input order: x, then params in dict order (each vn: Wf, Wd, g, b)
    const float* x     = (const float*)d_in[0];
    const float* c1Wf  = (const float*)d_in[1];
    const float* c1Wd  = (const float*)d_in[2];
    const float* c1g   = (const float*)d_in[3];
    const float* c1b   = (const float*)d_in[4];
    const float* c2Wf  = (const float*)d_in[5];
    const float* c2Wd  = (const float*)d_in[6];
    const float* c2g   = (const float*)d_in[7];
    const float* c2b   = (const float*)d_in[8];
    const float* c3Wf  = (const float*)d_in[9];
    const float* c3Wd  = (const float*)d_in[10];
    const float* c3g   = (const float*)d_in[11];
    const float* c3b   = (const float*)d_in[12];
    const float* c4Wf  = (const float*)d_in[13];
    const float* c4Wd  = (const float*)d_in[14];
    const float* c4g   = (const float*)d_in[15];
    const float* c4b   = (const float*)d_in[16];
    const float* c5Wf  = (const float*)d_in[17];
    const float* c5Wd  = (const float*)d_in[18];
    const float* c5g   = (const float*)d_in[19];
    const float* c5b   = (const float*)d_in[20];
    const float* s1Wf  = (const float*)d_in[21];
    const float* s1Wd  = (const float*)d_in[22];
    const float* s1g   = (const float*)d_in[23];
    const float* s1b   = (const float*)d_in[24];
    const float* s2Wf  = (const float*)d_in[25];
    const float* s2Wd  = (const float*)d_in[26];
    const float* s2g   = (const float*)d_in[27];
    const float* s2b   = (const float*)d_in[28];
    const float* slin  = (const float*)d_in[29];
    const float* t1W   = (const float*)d_in[30];
    const float* t2W   = (const float*)d_in[31];
    const float* t3W   = (const float*)d_in[32];
    const float* t3b   = (const float*)d_in[33];

    // workspace layout
    char* w = (char*)d_ws;
    auto alloc = [&](size_t bytes) { char* p = w; w += (bytes + 255) & ~(size_t)255; return p; };
    int*   idxb = (int*)  alloc((size_t)NBATCH * NPTS * KNBR * 4);
    float* xt   = (float*)alloc((size_t)NBATCH * NPTS * 128 * 4);
    float* x0b  = (float*)alloc((size_t)NBATCH * 3 * NPTS * 4);
    float* fcat = (float*)alloc((size_t)NBATCH * TOTF * 3 * NPTS * 4);
    float* gbuf = (float*)alloc((size_t)NBATCH * 170 * 3 * NPTS * 4);
    float* zbuf = (float*)alloc((size_t)NBATCH * 170 * 3 * 4);
    float* s1o  = (float*)alloc((size_t)NBATCH * 170 * 3 * NPTS * 4);
    float* s2o  = (float*)alloc((size_t)NBATCH * 85  * 3 * NPTS * 4);
    float* z0b  = (float*)alloc((size_t)NBATCH * 9 * NPTS * 4);
    float* qb   = (float*)alloc((size_t)NBATCH * 170 * 3 * NPTS * 4);
    float* db   = (float*)alloc((size_t)NBATCH * 170 * 3 * NPTS * 4);
    float* stats= (float*)alloc(4096);
    BF16*  XT0  = (BF16*) alloc((size_t)4096 * 1024 * 2);
    BF16*  XT1  = (BF16*) alloc((size_t)4096 * 1024 * 2);
    BF16*  XT2  = (BF16*) alloc((size_t)4096 * 1024 * 2);
    BF16*  W1b  = (BF16*) alloc((size_t)1024 * 1024 * 2);
    BF16*  W2b  = (BF16*) alloc((size_t)1024 * 1024 * 2);
    BF16*  W3b  = (BF16*) alloc((size_t)28928 * 1024 * 2);

    float* zout  = (float*)d_out;          // z  [B,170,3]
    float* theta = (float*)d_out + 2040;   // theta [B,170,170,N]

    const int PT = NBATCH * NPTS;          // 4096 points total
    const dim3 b256(256), bEdge(128);

    // ---- stage 1: knn on raw points, edge+vn c1 (cross), out ch [0,21)
    pack_x<<<(PT + 255) / 256, b256, 0, stream>>>(x, xt);
    x0fill<<<(PT + 255) / 256, b256, 0, stream>>>(x, x0b);
    knn_topk<<<NBATCH * (NPTS / 128), bEdge, 0, stream>>>(xt, 32, idxb);
    zero_u32<<<1, 64, 0, stream>>>((unsigned*)stats, 2 * 21);
    vn_edge_passA<1, 21, true><<<PT, bEdge, 0, stream>>>(x0b, 0, 1, idxb, c1Wf, stats);
    vn_edge_passB<1, 21, true><<<PT, bEdge, 0, stream>>>(x0b, 0, 1, idxb, c1Wf, c1Wd,
                                                         c1g, c1b, stats, fcat, 0, 0.0f);
    // ---- stage 2: knn on f1, c2, out ch [21,42)
    feat_to_xt<<<(PT + 255) / 256, b256, 0, stream>>>(fcat, 0, TOTF, 21, xt, 64);
    knn_topk<<<NBATCH * (NPTS / 128), bEdge, 0, stream>>>(xt, 64, idxb);
    zero_u32<<<1, 64, 0, stream>>>((unsigned*)stats, 2 * 21);
    vn_edge_passA<21, 21, false><<<PT, bEdge, 0, stream>>>(fcat, 0, TOTF, idxb, c2Wf, stats);
    vn_edge_passB<21, 21, false><<<PT, bEdge, 0, stream>>>(fcat, 0, TOTF, idxb, c2Wf, c2Wd,
                                                           c2g, c2b, stats, fcat, 21, 0.0f);
    // ---- stage 3: knn on f2, c3, out ch [42,84)
    feat_to_xt<<<(PT + 255) / 256, b256, 0, stream>>>(fcat, 21, TOTF, 21, xt, 64);
    knn_topk<<<NBATCH * (NPTS / 128), bEdge, 0, stream>>>(xt, 64, idxb);
    zero_u32<<<1, 128, 0, stream>>>((unsigned*)stats, 2 * 42);
    vn_edge_passA<21, 42, false><<<PT, bEdge, 0, stream>>>(fcat, 21, TOTF, idxb, c3Wf, stats);
    vn_edge_passB<21, 42, false><<<PT, bEdge, 0, stream>>>(fcat, 21, TOTF, idxb, c3Wf, c3Wd,
                                                           c3g, c3b, stats, fcat, 42, 0.0f);
    // ---- stage 4: knn on f3, c4, out ch [84,169)
    feat_to_xt<<<(PT + 255) / 256, b256, 0, stream>>>(fcat, 42, TOTF, 42, xt, 128);
    knn_topk<<<NBATCH * (NPTS / 128), bEdge, 0, stream>>>(xt, 128, idxb);
    zero_u32<<<1, 256, 0, stream>>>((unsigned*)stats, 2 * 85);
    vn_edge_passA<42, 85, false><<<PT, bEdge, 0, stream>>>(fcat, 42, TOTF, idxb, c4Wf, stats);
    vn_edge_passB<42, 85, false><<<PT, bEdge, 0, stream>>>(fcat, 42, TOTF, idxb, c4Wf, c4Wd,
                                                           c4g, c4b, stats, fcat, 84, 0.0f);
    // ---- c5 (shared dir, ns=0.2): fcat[169] -> g[170]
    zero_u32<<<2, 256, 0, stream>>>((unsigned*)stats, 2 * 170);
    vn_dense_passA<<<NBATCH * 170, b256, 0, stream>>>(fcat, nullptr, 0, 169, 169, 170, 1,
                                                      c5Wf, c5Wd, qb, db, stats);
    vn_dense_passB<<<2048, b256, 0, stream>>>(qb, db, stats, c5g, c5b, gbuf,
                                              170, 1, 0.2f, NBATCH * 170 * NPTS);
    // ---- global feature z
    zmean_kernel<<<8, b256, 0, stream>>>(gbuf, zbuf, zout);
    // ---- s1: h=[g; z] (340) -> 170
    zero_u32<<<2, 256, 0, stream>>>((unsigned*)stats, 2 * 170);
    vn_dense_passA<<<NBATCH * 170, b256, 0, stream>>>(gbuf, zbuf, 1, 340, 170, 170, 0,
                                                      s1Wf, s1Wd, qb, db, stats);
    vn_dense_passB<<<2048, b256, 0, stream>>>(qb, db, stats, s1g, s1b, s1o,
                                              170, 0, 0.2f, NBATCH * 170 * NPTS);
    // ---- s2: 170 -> 85
    zero_u32<<<1, 256, 0, stream>>>((unsigned*)stats, 2 * 85);
    vn_dense_passA<<<NBATCH * 85, b256, 0, stream>>>(s1o, nullptr, 0, 170, 170, 85, 0,
                                                     s2Wf, s2Wd, qb, db, stats);
    vn_dense_passB<<<1024, b256, 0, stream>>>(qb, db, stats, s2g, s2b, s2o,
                                              85, 0, 0.2f, NBATCH * 85 * NPTS);
    // ---- z0 and invariant features xs -> bf16 transposed activations
    z0_kernel<<<(NBATCH * 9 * NPTS + 255) / 256, b256, 0, stream>>>(s2o, slin, z0b);
    zero_u32<<<1024, b256, 0, stream>>>((unsigned*)XT0, (size_t)4096 * 1024 / 2);
    xs_kernel<<<(NBATCH * 340 * NPTS + 255) / 256, b256, 0, stream>>>(gbuf, zbuf, z0b, XT0);
    // ---- weight conversion
    cvt_w<<<2048, b256, 0, stream>>>(t1W, W1b, 1020, 1020, 1024, 1024);
    cvt_w<<<2048, b256, 0, stream>>>(t2W, W2b, 1020, 1020, 1024, 1024);
    cvt_w<<<8192, b256, 0, stream>>>(t3W, W3b, 28900, 1020, 28928, 1024);
    // ---- theta_net: WMMA GEMMs (128x128 block tiles, async LDS pipeline)
    {   // t1: 1024x1024 x 4096, relu -> XT1
        int blocks = (1024 / 128) * (4096 / 128);
        gemm_bf16<<<blocks, b256, 0, stream>>>(W1b, XT0, 1024, 1024, 4096, 0,
                                               XT1, nullptr, nullptr, 1020);
    }
    {   // t2 -> XT2
        int blocks = (1024 / 128) * (4096 / 128);
        gemm_bf16<<<blocks, b256, 0, stream>>>(W2b, XT1, 1024, 1024, 4096, 0,
                                               XT2, nullptr, nullptr, 1020);
    }
    {   // t3: 28928x1024 x 4096, +bias, fp32 scatter into theta [B,28900,N]
        int blocks = (28928 / 128) * (4096 / 128);
        gemm_bf16<<<blocks, b256, 0, stream>>>(W3b, XT2, 28928, 1024, 4096, 1,
                                               nullptr, theta, t3b, 28900);
    }
}
